// UTransformer_26225070310023
// MI455X (gfx1250) — matrix operand order repaired and checked by reading, not compiled, not run on hardware
//
#include <hip/hip_runtime.h>
#include <hip/hip_bf16.h>
#include <cstddef>
#include <cstdint>

// ---------------------------------------------------------------------------
// Universal Transformer (ACT) forward for MI455X / gfx1250.
// Dense GEMMs: V_WMMA_F32_16X16X32_F16 (f16 operands, f32 accum) — CDNA5's
// f32 WMMA is only 16x16x4, so the 16-bit matrix path gives 8x K-depth/op.
// Double-buffered LDS pipeline: A panels via GLOBAL_LOAD_ASYNC_TO_LDS_B128
// (ASYNCcnt DMA overlapped with WMMA), B panels loaded to registers early,
// converted/stored to LDS after compute. Fragments are pure ds_load_b128.
// ---------------------------------------------------------------------------

#define D_MODEL 512
#define DFF     2048
#define NVOCAB  32000
#define NNODES  512      // B*LS == B*LT
#define THRES   0.99f

typedef _Float16 v16h __attribute__((ext_vector_type(16)));
typedef _Float16 v8h  __attribute__((ext_vector_type(8)));
typedef float    v8f  __attribute__((ext_vector_type(8)));

// ---------------------------------------------------------------------------
// WMMA GEMM: C[M,N] = A[M,K] @ W[K,N] (+bias) (+relu), optional row mask so
// inactive ACT rows keep previous values (jnp.where semantics).
// Block tile 32x128 (8 waves, each 32x16 via two accumulators), K-chunk 64,
// double-buffered. M%32==0, N%128==0, K%128==0 hold for every call site.
// ---------------------------------------------------------------------------
__global__ __launch_bounds__(256) void k_gemm_f16(
    const float* __restrict__ A, const float* __restrict__ W,
    float* __restrict__ C, int M, int N, int K,
    const float* __restrict__ bias, const int* __restrict__ rowmask, int relu)
{
    __shared__ __align__(16) float    Af[2][32][68];    // A panels, f32 (async DMA)
    __shared__ __align__(16) _Float16 Bs[2][128][72];   // B panels, f16, transposed

    const int tid   = threadIdx.x;
    const int lane  = tid & 31;
    const int wave  = tid >> 5;
    const int mbase = blockIdx.y * 32;
    const int nblk  = blockIdx.x * 128;
    const int mrow  = lane & 15;
    const int hi    = lane >> 4;              // lanes 16-31: +8 in K, +8 in M
    const int ncol  = nblk + wave * 16 + mrow;

    v8f acc0 = {0.f,0.f,0.f,0.f,0.f,0.f,0.f,0.f};
    v8f acc1 = {0.f,0.f,0.f,0.f,0.f,0.f,0.f,0.f};

    float4 wreg[8];                           // in-flight B panel (next chunk)

    // --- staging helpers -------------------------------------------------
    auto asyncA = [&](int buf, int k0) {      // 32x64 f32, 16B per lane x2
        #pragma unroll
        for (int j = 0; j < 2; ++j) {
            int idx = tid + j * 256;
            int r  = idx >> 4;
            int c4 = (idx & 15) * 4;
            unsigned long long gsrc =
                (unsigned long long)(A + (mbase + r) * K + k0 + c4);
            unsigned ldst = (unsigned)(uintptr_t)(&Af[buf][r][c4]);
            asm volatile("global_load_async_to_lds_b128 %0, %1, off"
                         :: "v"(ldst), "v"(gsrc) : "memory");
        }
    };
    auto loadB = [&](int k0) {                // 64x128 coalesced float4 loads
        #pragma unroll
        for (int j = 0; j < 8; ++j) {
            int idx = tid + j * 256;
            int r  = idx >> 5;
            int c4 = (idx & 31) * 4;
            const float* wp = W + (k0 + r) * N + nblk + c4;
            if (k0 + 64 < K)                  // warm L2 one panel further ahead
                __builtin_prefetch(wp + 64 * N, 0, 0);
            wreg[j] = *(const float4*)wp;
        }
    };
    auto storeB = [&](int buf) {              // f16 transpose into LDS
        #pragma unroll
        for (int j = 0; j < 8; ++j) {
            int idx = tid + j * 256;
            int r  = idx >> 5;
            int c4 = (idx & 31) * 4;
            Bs[buf][c4 + 0][r] = (_Float16)wreg[j].x;
            Bs[buf][c4 + 1][r] = (_Float16)wreg[j].y;
            Bs[buf][c4 + 2][r] = (_Float16)wreg[j].z;
            Bs[buf][c4 + 3][r] = (_Float16)wreg[j].w;
        }
    };

    // --- prologue: stage panel 0 ----------------------------------------
    asyncA(0, 0);
    loadB(0);
    storeB(0);

    int p = 0;
    for (int k0 = 0; k0 < K; k0 += 64, p ^= 1) {
        asm volatile("s_wait_asynccnt 0x0" ::: "memory");
        __syncthreads();                      // panel p ready for all waves

        const bool more = (k0 + 64) < K;
        if (more) {                           // overlap next panel with WMMA
            asyncA(p ^ 1, k0 + 64);
            loadB(k0 + 64);
        }

        #pragma unroll
        for (int kk = 0; kk < 32 * 2; kk += 32) {
            // B fragment: two contiguous 8-half runs -> 2x ds_load_b128
            v8h blo = *(const v8h*)(&Bs[p][wave * 16 + mrow][kk + 8 * hi]);
            v8h bhi = *(const v8h*)(&Bs[p][wave * 16 + mrow][kk + 16 + 8 * hi]);
            v16h bf = __builtin_shufflevector(blo, bhi,
                0,1,2,3,4,5,6,7,8,9,10,11,12,13,14,15);

            // A fragments (f32 in LDS -> cvt to f16 at build)
            float4 a0 = *(const float4*)(&Af[p][mrow][kk + 8 * hi]);
            float4 a1 = *(const float4*)(&Af[p][mrow][kk + 8 * hi + 4]);
            float4 a2 = *(const float4*)(&Af[p][mrow][kk + 16 + 8 * hi]);
            float4 a3 = *(const float4*)(&Af[p][mrow][kk + 16 + 8 * hi + 4]);
            v16h af0;
            af0[0]=(_Float16)a0.x; af0[1]=(_Float16)a0.y; af0[2]=(_Float16)a0.z; af0[3]=(_Float16)a0.w;
            af0[4]=(_Float16)a1.x; af0[5]=(_Float16)a1.y; af0[6]=(_Float16)a1.z; af0[7]=(_Float16)a1.w;
            af0[8]=(_Float16)a2.x; af0[9]=(_Float16)a2.y; af0[10]=(_Float16)a2.z; af0[11]=(_Float16)a2.w;
            af0[12]=(_Float16)a3.x; af0[13]=(_Float16)a3.y; af0[14]=(_Float16)a3.z; af0[15]=(_Float16)a3.w;

            float4 b0 = *(const float4*)(&Af[p][mrow + 16][kk + 8 * hi]);
            float4 b1 = *(const float4*)(&Af[p][mrow + 16][kk + 8 * hi + 4]);
            float4 b2 = *(const float4*)(&Af[p][mrow + 16][kk + 16 + 8 * hi]);
            float4 b3 = *(const float4*)(&Af[p][mrow + 16][kk + 16 + 8 * hi + 4]);
            v16h af1;
            af1[0]=(_Float16)b0.x; af1[1]=(_Float16)b0.y; af1[2]=(_Float16)b0.z; af1[3]=(_Float16)b0.w;
            af1[4]=(_Float16)b1.x; af1[5]=(_Float16)b1.y; af1[6]=(_Float16)b1.z; af1[7]=(_Float16)b1.w;
            af1[8]=(_Float16)b2.x; af1[9]=(_Float16)b2.y; af1[10]=(_Float16)b2.z; af1[11]=(_Float16)b2.w;
            af1[12]=(_Float16)b3.x; af1[13]=(_Float16)b3.y; af1[14]=(_Float16)b3.z; af1[15]=(_Float16)b3.w;

            acc0 = __builtin_amdgcn_wmma_f32_16x16x32_f16(
                false, af0, false, bf, (short)0, acc0, false, false);
            acc1 = __builtin_amdgcn_wmma_f32_16x16x32_f16(
                false, af1, false, bf, (short)0, acc1, false, false);
        }

        if (more) storeB(p ^ 1);              // cvt+store after compute issue
    }

    const float bb = bias ? bias[ncol] : 0.f;
    const int mo = hi * 8;                    // C layout: lanes>=16 hold M=r+8
    #pragma unroll
    for (int r = 0; r < 8; ++r) {
        int m0 = mbase + r + mo;
        if (!rowmask || rowmask[m0]) {
            float v = acc0[r] + bb;
            if (relu) v = fmaxf(v, 0.f);
            C[m0 * N + ncol] = v;
        }
        int m1 = m0 + 16;
        if (!rowmask || rowmask[m1]) {
            float v = acc1[r] + bb;
            if (relu) v = fmaxf(v, 0.f);
            C[m1 * N + ncol] = v;
        }
    }
}

// ---------------------------------------------------------------------------
// Graph attention: one wave per (dst node, head).
// mode 0: encoder self (all 64), 1: decoder causal (<= pos), 2: cross.
// ---------------------------------------------------------------------------
__global__ __launch_bounds__(256) void k_attn(
    const float* __restrict__ Q, const float* __restrict__ K,
    const float* __restrict__ V, float* __restrict__ OUT, int mode)
{
    int wid  = (blockIdx.x * 256 + threadIdx.x) >> 5;
    int lane = threadIdx.x & 31;
    int d = wid >> 3;
    int h = wid & 7;
    int base = (d >> 6) << 6;
    int nsrc = (mode == 1) ? ((d & 63) + 1) : 64;

    const float* qp = Q + d * D_MODEL + h * 64;
    float q0 = qp[lane], q1 = qp[lane + 32];
    float z = 0.f, w0 = 0.f, w1 = 0.f;

    for (int t = 0; t < nsrc; ++t) {
        const float* kp = K + (base + t) * D_MODEL + h * 64;
        float p = q0 * kp[lane] + q1 * kp[lane + 32];
        for (int off = 16; off; off >>= 1) p += __shfl_xor(p, off, 32);
        float sc = fminf(10.f, fmaxf(-10.f, p * 0.125f));
        float e = __expf(sc);
        const float* vp = V + (base + t) * D_MODEL + h * 64;
        z += e;
        w0 += e * vp[lane];
        w1 += e * vp[lane + 32];
    }
    float zz = (z > 0.f) ? z : 1.f;
    OUT[d * D_MODEL + h * 64 + lane]      = w0 / zz;
    OUT[d * D_MODEL + h * 64 + lane + 32] = w1 / zz;
}

// ---------------------------------------------------------------------------
// Per-node elementwise / LN / ACT kernels (block == one node of 512 dims)
// ---------------------------------------------------------------------------
__global__ __launch_bounds__(256) void k_ln(
    const float* __restrict__ X, const float* __restrict__ g,
    const float* __restrict__ b, float* __restrict__ Y)
{
    __shared__ float red[256];
    int n = blockIdx.x, t = threadIdx.x;
    const float* x = X + n * D_MODEL;
    float x0 = x[t], x1 = x[t + 256];
    red[t] = x0 + x1;
    __syncthreads();
    for (int off = 128; off; off >>= 1) { if (t < off) red[t] += red[t + off]; __syncthreads(); }
    float mean = red[0] * (1.f / 512.f);
    __syncthreads();
    float d0 = x0 - mean, d1 = x1 - mean;
    red[t] = d0 * d0 + d1 * d1;
    __syncthreads();
    for (int off = 128; off; off >>= 1) { if (t < off) red[t] += red[t + off]; __syncthreads(); }
    float inv = rsqrtf(red[0] * (1.f / 512.f) + 1e-5f);
    Y[n * D_MODEL + t]       = d0 * inv * g[t] + b[t];
    Y[n * D_MODEL + t + 256] = d1 * inv * g[t + 256] + b[t + 256];
}

__global__ __launch_bounds__(256) void k_halt(
    const float* __restrict__ X, const float* __restrict__ g,
    const float* __restrict__ b, const float* __restrict__ Wv,
    const float* __restrict__ bias, float* __restrict__ PH)
{
    __shared__ float red[256];
    int n = blockIdx.x, t = threadIdx.x;
    const float* x = X + n * D_MODEL;
    float x0 = x[t], x1 = x[t + 256];
    red[t] = x0 + x1;
    __syncthreads();
    for (int off = 128; off; off >>= 1) { if (t < off) red[t] += red[t + off]; __syncthreads(); }
    float mean = red[0] * (1.f / 512.f);
    __syncthreads();
    float d0 = x0 - mean, d1 = x1 - mean;
    red[t] = d0 * d0 + d1 * d1;
    __syncthreads();
    for (int off = 128; off; off >>= 1) { if (t < off) red[t] += red[t + off]; __syncthreads(); }
    float inv = rsqrtf(red[0] * (1.f / 512.f) + 1e-5f);
    __syncthreads();
    red[t] = (d0 * inv * g[t] + b[t]) * Wv[t]
           + (d1 * inv * g[t + 256] + b[t + 256]) * Wv[t + 256];
    __syncthreads();
    for (int off = 128; off; off >>= 1) { if (t < off) red[t] += red[t + off]; __syncthreads(); }
    if (t == 0) PH[n] = 1.f / (1.f + __expf(-(red[0] + bias[0])));
}

__global__ __launch_bounds__(256) void k_pre(
    float* __restrict__ X, const int* __restrict__ ACT, int* __restrict__ STP,
    const float* __restrict__ pos, const float* __restrict__ timetab)
{
    int n = blockIdx.x, t = threadIdx.x;
    if (!ACT[n]) return;
    int st = STP[n];
    int base = n * D_MODEL;
    int pn = n & 63;
    X[base + t]       += pos[pn * D_MODEL + t]       + timetab[st * D_MODEL + t];
    X[base + t + 256] += pos[pn * D_MODEL + t + 256] + timetab[st * D_MODEL + t + 256];
    __syncthreads();
    if (t == 0) STP[n] = st + 1;
}

__global__ void k_embed(float* __restrict__ X, const int* __restrict__ tok,
                        const float* __restrict__ emb)
{
    int idx = blockIdx.x * 256 + threadIdx.x;
    int n = idx >> 9, d = idx & 511;
    X[idx] = emb[tok[n] * D_MODEL + d];
}

__global__ void k_zero(float* __restrict__ X)
{
    X[blockIdx.x * 256 + threadIdx.x] = 0.f;
}

__global__ void k_init(float* SUMP, float* R, int* ACT, int* STP)
{
    int n = blockIdx.x * 256 + threadIdx.x;
    if (n < NNODES) { SUMP[n] = 0.f; R[n] = 1.f; ACT[n] = 1; STP[n] = 0; }
}

__global__ void k_resid(float* __restrict__ X, const float* __restrict__ O,
                        const int* __restrict__ ACT)
{
    int idx = blockIdx.x * 256 + threadIdx.x;
    if (ACT[idx >> 9]) X[idx] += O[idx];
}

// s += ((1-c)*r_old + c*ph) * x  on previously-active rows (reads OLD state)
__global__ void k_act_s(const float* __restrict__ X, const float* __restrict__ PH,
                        const float* __restrict__ SUMP, const float* __restrict__ R,
                        const int* __restrict__ ACT, float* __restrict__ S, int end)
{
    int idx = blockIdx.x * 256 + threadIdx.x;
    int n = idx >> 9;
    if (!ACT[n]) return;
    float ph  = PH[n];
    float sp1 = SUMP[n] + ph;
    int act1  = (sp1 < THRES) && !end;
    float coef = act1 ? ph : R[n];
    S[idx] += coef * X[idx];
}

__global__ void k_act_state(const float* __restrict__ PH, float* __restrict__ SUMP,
                            float* __restrict__ R, int* __restrict__ ACT, int end)
{
    int n = blockIdx.x * 256 + threadIdx.x;
    if (n >= NNODES || !ACT[n]) return;
    float sp1 = SUMP[n] + PH[n];
    int act1  = (sp1 < THRES) && !end;
    if (act1) R[n] = 1.f - sp1;
    SUMP[n] = sp1;
    ACT[n]  = act1;
}

__global__ __launch_bounds__(256) void k_lsm(float* __restrict__ L)
{
    __shared__ float red[256];
    int n = blockIdx.x, t = threadIdx.x;
    float* row = L + n * NVOCAB;
    float m = -3.0e38f;
    for (int c = t; c < NVOCAB; c += 256) m = fmaxf(m, row[c]);
    red[t] = m;
    __syncthreads();
    for (int off = 128; off; off >>= 1) { if (t < off) red[t] = fmaxf(red[t], red[t + off]); __syncthreads(); }
    float M = red[0];
    __syncthreads();
    float s = 0.f;
    for (int c = t; c < NVOCAB; c += 256) s += __expf(row[c] - M);
    red[t] = s;
    __syncthreads();
    for (int off = 128; off; off >>= 1) { if (t < off) red[t] += red[t + off]; __syncthreads(); }
    float lse = M + __logf(red[0]);
    for (int c = t; c < NVOCAB; c += 256) row[c] -= lse;
}

__global__ __launch_bounds__(256) void k_loss(const float* __restrict__ RE,
                                              const float* __restrict__ RD,
                                              float* __restrict__ out)
{
    __shared__ float red[256];
    int t = threadIdx.x;
    red[t] = RE[t] + RE[t + 256] + RD[t] + RD[t + 256];
    __syncthreads();
    for (int off = 128; off; off >>= 1) { if (t < off) red[t] += red[t + off]; __syncthreads(); }
    if (t == 0) out[0] = red[0] * (0.01f / 1024.f);
}

// ---------------------------------------------------------------------------
// Host-side orchestration.
// params flattened in setup_inputs() insertion order, starting at d_in[8]:
// ---------------------------------------------------------------------------
enum {
    P_EMB_SRC = 8, P_EMB_TGT, P_POS, P_TIME,
    P_E_LN_G, P_E_LN_B, P_E_WQ, P_E_WK, P_E_WV, P_E_WO, P_E_BO,
    P_D_LN_G, P_D_LN_B, P_D_WQ, P_D_WK, P_D_WV, P_D_WO, P_D_BO,
    P_C_LN_G, P_C_LN_B, P_C_WQ, P_C_WK, P_C_WV, P_C_WO, P_C_BO,
    P_E_FFLN_G, P_E_FFLN_B, P_E_W1, P_E_B1, P_E_W2, P_E_B2,
    P_C_FFLN_G, P_C_FFLN_B, P_C_W1, P_C_B1, P_C_W2, P_C_B2,
    P_HE_LN_G, P_HE_LN_B, P_HE_W, P_HE_B,
    P_HD_LN_G, P_HD_LN_B, P_HD_W, P_HD_B,
    P_ENORM_G, P_ENORM_B, P_DNORM_G, P_DNORM_B,
    P_GW, P_GB
};

extern "C" void kernel_launch(void* const* d_in, const int* in_sizes, int n_in,
                              void* d_out, int out_size, void* d_ws, size_t ws_size,
                              hipStream_t stream)
{
    (void)in_sizes; (void)n_in; (void)out_size; (void)ws_size;
    #define PF(i) ((const float*)d_in[(i)])

    const size_t F = (size_t)NNODES * D_MODEL;   // 262144 floats
    float* ws   = (float*)d_ws;
    float* X    = ws;
    float* XN   = ws + 1 * F;
    float* Qb   = ws + 2 * F;
    float* Kb   = ws + 3 * F;
    float* Vb   = ws + 4 * F;
    float* QC   = ws + 5 * F;
    float* AT   = ws + 6 * F;
    float* O    = ws + 7 * F;
    float* S    = ws + 8 * F;
    float* XENC = ws + 9 * F;
    float* KC   = ws + 10 * F;
    float* VC   = ws + 11 * F;
    float* FFH  = ws + 12 * F;   // [512, 2048] = 4F
    float* PH   = ws + 16 * F;
    float* SUMP = PH + NNODES;
    float* R_E  = SUMP + NNODES;
    float* R_D  = R_E + NNODES;
    int*   ACT  = (int*)(R_D + NNODES);
    int*   STP  = ACT + NNODES;

    float* logits = (float*)d_out;                       // [512, 32000]
    float* loss   = logits + (size_t)NNODES * NVOCAB;    // scalar

    auto gemm = [&](const float* A, const float* W, float* C, int M, int N, int K,
                    const float* bias, const int* mask, int relu) {
        dim3 grid(N / 128, M / 32);
        k_gemm_f16<<<grid, 256, 0, stream>>>(A, W, C, M, N, K, bias, mask, relu);
    };

    // ------------------------- encoder ACT loop -------------------------
    k_embed<<<1024, 256, 0, stream>>>(X, (const int*)d_in[0], PF(P_EMB_SRC));
    k_zero<<<1024, 256, 0, stream>>>(S);
    k_init<<<2, 256, 0, stream>>>(SUMP, R_E, ACT, STP);
    for (int step = 0; step < 8; ++step) {
        int end = (step == 7);
        k_pre<<<NNODES, 256, 0, stream>>>(X, ACT, STP, PF(P_POS), PF(P_TIME));
        k_ln<<<NNODES, 256, 0, stream>>>(X, PF(P_E_LN_G), PF(P_E_LN_B), XN);
        gemm(XN, PF(P_E_WQ), Qb, NNODES, D_MODEL, D_MODEL, nullptr, ACT, 0);
        gemm(XN, PF(P_E_WK), Kb, NNODES, D_MODEL, D_MODEL, nullptr, ACT, 0);
        gemm(XN, PF(P_E_WV), Vb, NNODES, D_MODEL, D_MODEL, nullptr, ACT, 0);
        k_attn<<<NNODES, 256, 0, stream>>>(Qb, Kb, Vb, AT, 0);
        gemm(AT, PF(P_E_WO), O, NNODES, D_MODEL, D_MODEL, PF(P_E_BO), nullptr, 0);
        k_resid<<<1024, 256, 0, stream>>>(X, O, ACT);
        k_ln<<<NNODES, 256, 0, stream>>>(X, PF(P_E_FFLN_G), PF(P_E_FFLN_B), XN);
        gemm(XN, PF(P_E_W1), FFH, NNODES, DFF, D_MODEL, PF(P_E_B1), nullptr, 1);
        gemm(FFH, PF(P_E_W2), O, NNODES, D_MODEL, DFF, PF(P_E_B2), nullptr, 0);
        k_resid<<<1024, 256, 0, stream>>>(X, O, ACT);
        k_halt<<<NNODES, 256, 0, stream>>>(X, PF(P_HE_LN_G), PF(P_HE_LN_B),
                                           PF(P_HE_W), PF(P_HE_B), PH);
        k_act_s<<<1024, 256, 0, stream>>>(X, PH, SUMP, R_E, ACT, S, end);
        k_act_state<<<2, 256, 0, stream>>>(PH, SUMP, R_E, ACT, end);
    }
    k_ln<<<NNODES, 256, 0, stream>>>(S, PF(P_ENORM_G), PF(P_ENORM_B), XENC);

    // cross-attn keys/values from final encoder states
    k_ln<<<NNODES, 256, 0, stream>>>(XENC, PF(P_C_LN_G), PF(P_C_LN_B), XN);
    gemm(XN, PF(P_C_WK), KC, NNODES, D_MODEL, D_MODEL, nullptr, nullptr, 0);
    gemm(XN, PF(P_C_WV), VC, NNODES, D_MODEL, D_MODEL, nullptr, nullptr, 0);

    // ------------------------- decoder ACT loop -------------------------
    k_embed<<<1024, 256, 0, stream>>>(X, (const int*)d_in[1], PF(P_EMB_TGT));
    k_zero<<<1024, 256, 0, stream>>>(S);
    k_init<<<2, 256, 0, stream>>>(SUMP, R_D, ACT, STP);
    for (int step = 0; step < 8; ++step) {
        int end = (step == 7);
        k_pre<<<NNODES, 256, 0, stream>>>(X, ACT, STP, PF(P_POS), PF(P_TIME));
        k_ln<<<NNODES, 256, 0, stream>>>(X, PF(P_D_LN_G), PF(P_D_LN_B), XN);
        gemm(XN, PF(P_D_WQ), Qb, NNODES, D_MODEL, D_MODEL, nullptr, ACT, 0);
        gemm(XN, PF(P_D_WK), Kb, NNODES, D_MODEL, D_MODEL, nullptr, ACT, 0);
        gemm(XN, PF(P_D_WV), Vb, NNODES, D_MODEL, D_MODEL, nullptr, ACT, 0);
        k_attn<<<NNODES, 256, 0, stream>>>(Qb, Kb, Vb, AT, 1);       // causal
        gemm(AT, PF(P_D_WO), O, NNODES, D_MODEL, D_MODEL, PF(P_D_BO), nullptr, 0);
        k_resid<<<1024, 256, 0, stream>>>(X, O, ACT);
        k_ln<<<NNODES, 256, 0, stream>>>(X, PF(P_C_LN_G), PF(P_C_LN_B), XN);
        gemm(XN, PF(P_C_WQ), QC, NNODES, D_MODEL, D_MODEL, nullptr, ACT, 0);
        k_attn<<<NNODES, 256, 0, stream>>>(QC, KC, VC, AT, 2);       // cross
        gemm(AT, PF(P_C_WO), O, NNODES, D_MODEL, D_MODEL, PF(P_C_BO), nullptr, 0);
        k_resid<<<1024, 256, 0, stream>>>(X, O, ACT);
        k_ln<<<NNODES, 256, 0, stream>>>(X, PF(P_C_FFLN_G), PF(P_C_FFLN_B), XN);
        gemm(XN, PF(P_C_W1), FFH, NNODES, DFF, D_MODEL, PF(P_C_B1), nullptr, 1);
        gemm(FFH, PF(P_C_W2), O, NNODES, D_MODEL, DFF, PF(P_C_B2), nullptr, 0);
        k_resid<<<1024, 256, 0, stream>>>(X, O, ACT);
        k_halt<<<NNODES, 256, 0, stream>>>(X, PF(P_HD_LN_G), PF(P_HD_LN_B),
                                           PF(P_HD_W), PF(P_HD_B), PH);
        k_act_s<<<1024, 256, 0, stream>>>(X, PH, SUMP, R_D, ACT, S, end);
        k_act_state<<<2, 256, 0, stream>>>(PH, SUMP, R_D, ACT, end);
    }

    // ------------------------- output head -------------------------
    k_ln<<<NNODES, 256, 0, stream>>>(S, PF(P_DNORM_G), PF(P_DNORM_B), XN);
    gemm(XN, PF(P_GW), logits, NNODES, NVOCAB, D_MODEL, PF(P_GB), nullptr, 0);
    k_lsm<<<NNODES, 256, 0, stream>>>(logits);
    k_loss<<<1, 256, 0, stream>>>(R_E, R_D, loss);

    #undef PF
}
